// PERC_27075473834508
// MI455X (gfx1250) — matrix operand order repaired and checked by reading, not compiled, not run on hardware
//
#include <hip/hip_runtime.h>

// ---------------------------------------------------------------------------
// CDNA5 / gfx1250 implementation of the RGAT+MHA+GRU reference.
// All large GEMMs + both attention softmax-aggregations run on
// v_wmma_f32_16x16x32_f16 (wave32, 16x16 tiles, f32 accumulate).
// GEMM: 4-wave blocks, double-buffered LDS staging so next-chunk global
// loads overlap current-chunk WMMAs. Attention operands come from
// pre-transposed f16 buffers so fragment builds are 128-bit loads.
// ---------------------------------------------------------------------------

typedef __attribute__((ext_vector_type(16))) _Float16 v16h;
typedef __attribute__((ext_vector_type(8)))  _Float16 v8h;
typedef __attribute__((ext_vector_type(8)))  float    v8f;

#define WMMA_F16(a, b, c) \
  __builtin_amdgcn_wmma_f32_16x16x32_f16(false, (a), false, (b), (short)0, (c), false, false)

__device__ __forceinline__ v8f vzero8() {
  v8f z;
#pragma unroll
  for (int i = 0; i < 8; ++i) z[i] = 0.f;
  return z;
}

// A-fragment (16-bit, 16x32, ISA 7.12.2): lane-half hi, packed-half j.
// j in [0,8)  -> k = 8*hi + j          (contiguous 8)
// j in [8,16) -> k = 16 + 8*hi + (j-8) (contiguous 8)
__device__ __forceinline__ int a_k(int hi, int j) {
  return (j < 8) ? (8 * hi + j) : (16 + 8 * hi + (j - 8));
}

// ---------------------------------------------------------------------------
// WMMA GEMM: C[Mrows,Ncols] = act(A@W + bias + addsrc)
// 128 threads (4 waves) per block, 64x64 tile, double-buffered LDS f16 tiles.
// Mrows % 64 == 0, Kdim % 32 == 0. act: 0 none, 1 relu.
// ---------------------------------------------------------------------------
__global__ void gemm_wmma_kernel(const float* __restrict__ A,
                                 const float* __restrict__ W,
                                 const float* __restrict__ bias,
                                 const float* __restrict__ addsrc,
                                 float* __restrict__ C,
                                 int Mrows, int Kdim, int Ncols, int act) {
  __shared__ _Float16 sA[2][64][32];  // [buf][m][k]
  __shared__ _Float16 sW[2][64][32];  // [buf][n][k] (transposed at staging)

  const int tid = threadIdx.x;  // 0..127
  const int wvid = tid >> 5;    // wave id: rows 16*wvid..16*wvid+15
  const int lane = tid & 31;
  const int r = lane & 15, hi = lane >> 4;
  const int mbase = blockIdx.x * 64;
  const int nbase = blockIdx.y * 64;
  const bool ncVec = ((Ncols & 3) == 0);

  v8f acc[4];
#pragma unroll
  for (int t = 0; t < 4; ++t) acc[t] = vzero8();

  float4 aq[4], wq4[4];  // in-flight staging registers (distinct quads)

  auto load_chunk = [&](int k0) {
#pragma unroll
    for (int i = 0; i < 4; ++i) {
      const int e = i * 512 + tid * 4;
      const int row = e >> 5, col = e & 31;
      const float* ap = A + (long)(mbase + row) * Kdim + k0 + col;
      __builtin_prefetch(ap + 64, 0, 3);  // k+2 chunk -> global_prefetch_b8
      aq[i] = *(const float4*)ap;
    }
#pragma unroll
    for (int i = 0; i < 4; ++i) {
      const int e = i * 512 + tid * 4;
      const int row = e >> 6, col = e & 63;  // row = k-local, col = n-local
      const int n = nbase + col;
      if (ncVec && (n + 3 < Ncols)) {
        wq4[i] = *(const float4*)(W + (long)(k0 + row) * Ncols + n);
      } else {
        float4 t4;
        t4.x = (n + 0 < Ncols) ? W[(long)(k0 + row) * Ncols + n + 0] : 0.f;
        t4.y = (n + 1 < Ncols) ? W[(long)(k0 + row) * Ncols + n + 1] : 0.f;
        t4.z = (n + 2 < Ncols) ? W[(long)(k0 + row) * Ncols + n + 2] : 0.f;
        t4.w = (n + 3 < Ncols) ? W[(long)(k0 + row) * Ncols + n + 3] : 0.f;
        wq4[i] = t4;
      }
    }
  };

  auto store_chunk = [&](int buf) {
#pragma unroll
    for (int i = 0; i < 4; ++i) {
      const int e = i * 512 + tid * 4;
      const int row = e >> 5, col = e & 31;
      sA[buf][row][col + 0] = (_Float16)aq[i].x;
      sA[buf][row][col + 1] = (_Float16)aq[i].y;
      sA[buf][row][col + 2] = (_Float16)aq[i].z;
      sA[buf][row][col + 3] = (_Float16)aq[i].w;
    }
#pragma unroll
    for (int i = 0; i < 4; ++i) {
      const int e = i * 512 + tid * 4;
      const int row = e >> 6, col = e & 63;
      sW[buf][col + 0][row] = (_Float16)wq4[i].x;
      sW[buf][col + 1][row] = (_Float16)wq4[i].y;
      sW[buf][col + 2][row] = (_Float16)wq4[i].z;
      sW[buf][col + 3][row] = (_Float16)wq4[i].w;
    }
  };

  // prologue: stage chunk 0 into buffer 0
  load_chunk(0);
  store_chunk(0);

#pragma unroll 1
  for (int k0 = 0; k0 < Kdim; k0 += 32) {
    const int cur = (k0 >> 5) & 1;
    const bool more = (k0 + 32) < Kdim;
    if (more) load_chunk(k0 + 32);  // loads fly under the WMMAs below
    __syncthreads();                // cur buffer now fully staged

    v16h af;
    {
      const v8h a0 = *(const v8h*)&sA[cur][wvid * 16 + r][8 * hi];
      const v8h a1 = *(const v8h*)&sA[cur][wvid * 16 + r][16 + 8 * hi];
#pragma unroll
      for (int j = 0; j < 8; ++j) { af[j] = a0[j]; af[j + 8] = a1[j]; }
    }
#pragma unroll
    for (int t = 0; t < 4; ++t) {
      v16h bf;
      const v8h b0 = *(const v8h*)&sW[cur][t * 16 + r][8 * hi];
      const v8h b1 = *(const v8h*)&sW[cur][t * 16 + r][16 + 8 * hi];
#pragma unroll
      for (int j = 0; j < 8; ++j) { bf[j] = b0[j]; bf[j + 8] = b1[j]; }
      acc[t] = WMMA_F16(af, bf, acc[t]);
    }
    if (more) store_chunk(cur ^ 1);  // write other buffer; no race with readers
  }

#pragma unroll
  for (int v = 0; v < 8; ++v) {
    const int orow = mbase + wvid * 16 + v + 8 * hi;
#pragma unroll
    for (int t = 0; t < 4; ++t) {
      const int col = nbase + t * 16 + r;
      if (col < Ncols) {
        float x = acc[t][v];
        if (bias)   x += bias[col];
        if (addsrc) x += addsrc[(long)orow * Ncols + col];
        if (act == 1) x = fmaxf(x, 0.f);
        C[(long)orow * Ncols + col] = x;
      }
    }
  }
}

// ---------------------------------------------------------------------------
// f32 [B,N,HID] -> f16 [B,HID,N]  (for GAT B-fragments, contiguous in k)
// ---------------------------------------------------------------------------
__global__ void h_transpose_kernel(const float* __restrict__ X,
                                   _Float16* __restrict__ XT) {
  const int tid = blockIdx.x * blockDim.x + threadIdx.x;  // B*N*HID
  const int b = tid >> 18;
  const int n = (tid >> 8) & 1023;
  const int d = tid & 255;
  XT[((long)b * 256 + d) * 1024 + n] = (_Float16)X[tid];
}

// f32 [32,1024,64] -> f16 [32,64,1024] (for MHA P@V B-fragments)
__global__ void v_transpose_kernel(const float* __restrict__ X,
                                   _Float16* __restrict__ XT) {
  const int tid = blockIdx.x * blockDim.x + threadIdx.x;  // 32*1024*64
  const int bh = tid >> 16;
  const int s = (tid >> 6) & 1023;
  const int d = tid & 63;
  XT[((long)bh * 64 + d) * 1024 + s] = (_Float16)X[tid];
}

// ---------------------------------------------------------------------------
// Fused RGAT attention + aggregation (flash-style online softmax):
// Agg[b,q,:] = elu( softmax_k( mask(lrelu(s1[q]+s2[k]+rel[s_mask])) ) @ h )
// One wave per (b, 16-query tile); 16x256 f32 accumulators, WMMA P@h.
// ---------------------------------------------------------------------------
__global__ void gat_agg_kernel(const _Float16* __restrict__ hT,  // [B,HID,N] f16
                               const float* __restrict__ s1,     // [B,N]
                               const float* __restrict__ s2,     // [B,N]
                               const int* __restrict__ smask,    // [B,N,N]
                               const int* __restrict__ adjm,     // [B,N,N]
                               const float* __restrict__ relsc,  // [2]
                               float* __restrict__ Agg) {        // [B,N,HID]
  const int wid = blockIdx.x;  // B * 64
  const int b = wid >> 6;
  const int qt = wid & 63;
  const int lane = threadIdx.x & 31;
  const int r = lane & 15, hi = lane >> 4;
  const int q = qt * 16 + r;

  __shared__ float sM[16], sAl[16], sL[16], sRed[32];

  v8f acc[16];
#pragma unroll
  for (int t = 0; t < 16; ++t) acc[t] = vzero8();
  if (hi == 0) { sM[r] = -3.0e38f; sL[r] = 0.f; }
  __syncthreads();

  const float rel0 = relsc[0], rel1 = relsc[1];
  const float s1q = s1[b * 1024 + q];
  const float* s2b = s2 + b * 1024;
  const long rowoff = ((long)b * 1024 + q) * 1024;
  const int* mrow = smask + rowoff;
  const int* arow = adjm + rowoff;
  const _Float16* hTb = hT + (long)b * 262144;  // 256*1024

#pragma unroll 1
  for (int k0 = 0; k0 < 1024; k0 += 32) {
    float ev[16];
#pragma unroll
    for (int j = 0; j < 16; ++j) {
      const int k = k0 + a_k(hi, j);
      float e = s1q + s2b[k] + ((mrow[k] != 0) ? rel1 : rel0);
      e = (e > 0.f) ? e : 0.1f * e;    // leaky_relu(0.1)
      if (arow[k] == 0) e = -9.0e15f;  // masked fill
      ev[j] = e;
    }
    float lm = ev[0];
#pragma unroll
    for (int j = 1; j < 16; ++j) lm = fmaxf(lm, ev[j]);
    sRed[lane] = lm;
    __syncthreads();
    if (hi == 0) {
      const float mo = sM[r];
      const float mn = fmaxf(mo, fmaxf(sRed[r], sRed[r + 16]));
      sM[r] = mn;
      sAl[r] = __expf(mo - mn);
    }
    __syncthreads();
    const float mn = sM[r];
    v16h pf;
    float ls = 0.f;
#pragma unroll
    for (int j = 0; j < 16; ++j) {
      const float p = __expf(ev[j] - mn);
      ls += p;
      pf[j] = (_Float16)p;
    }
#pragma unroll
    for (int v = 0; v < 8; ++v) {
      const float al = sAl[v + 8 * hi];
#pragma unroll
      for (int t = 0; t < 16; ++t) acc[t][v] *= al;
    }
    sRed[lane] = ls;
    __syncthreads();
    if (hi == 0) sL[r] = sL[r] * sAl[r] + sRed[r] + sRed[r + 16];
    // P (16x32 probs) @ h-tile (32x256): B-frags are contiguous f16 runs in hT
#pragma unroll
    for (int t = 0; t < 16; ++t) {
      const _Float16* cp = hTb + ((long)(t * 16 + r) << 10) + k0 + 8 * hi;
      const v8h b0 = *(const v8h*)cp;
      const v8h b1 = *(const v8h*)(cp + 16);
      v16h bf;
#pragma unroll
      for (int j = 0; j < 8; ++j) { bf[j] = b0[j]; bf[j + 8] = b1[j]; }
      acc[t] = WMMA_F16(pf, bf, acc[t]);
    }
    __syncthreads();
  }

#pragma unroll
  for (int v = 0; v < 8; ++v) {
    const float li = 1.f / fmaxf(sL[v + 8 * hi], 1e-30f);
    const int row = qt * 16 + v + 8 * hi;
#pragma unroll
    for (int t = 0; t < 16; ++t) {
      float xv = acc[t][v] * li;
      xv = (xv > 0.f) ? xv : (__expf(xv) - 1.f);  // elu
      Agg[((long)b * 1024 + row) * 256 + t * 16 + r] = xv;
    }
  }
}

// ---------------------------------------------------------------------------
// Fused MHA (flash-style). The reference's reshape(B*NH,-1,DH) is a flat
// reinterpretation: Q/K/V/ctx are [32, 1024, 64] views of [B,N,HID] buffers.
// One wave per (head, 16-query tile). V comes pre-transposed in f16.
// ---------------------------------------------------------------------------
__global__ void mha_kernel(const float* __restrict__ Qb,
                           const float* __restrict__ Kb,
                           const _Float16* __restrict__ vT,  // [32,64,1024] f16
                           float* __restrict__ ctx) {
  const int wid = blockIdx.x;  // 32 heads * 64 tiles
  const int bh = wid >> 6;
  const int st = wid & 63;
  const int lane = threadIdx.x & 31;
  const int r = lane & 15, hi = lane >> 4;
  const long base = (long)bh * 1024 * 64;

  __shared__ float sM[16], sAl[16], sL[16];
  __shared__ float sS[16][33];

  if (hi == 0) { sM[r] = -3.0e38f; sL[r] = 0.f; }

  // Q fragments: contiguous 8-float runs -> float4 loads
  v16h qf[2];
  const int sq = st * 16 + r;
#pragma unroll
  for (int c = 0; c < 2; ++c) {
    const float* qp = Qb + base + (long)sq * 64 + c * 32 + 8 * hi;
    const float4 q0 = *(const float4*)qp;
    const float4 q1 = *(const float4*)(qp + 4);
    const float4 q2 = *(const float4*)(qp + 16);
    const float4 q3 = *(const float4*)(qp + 20);
    qf[c][0] = (_Float16)q0.x;  qf[c][1] = (_Float16)q0.y;
    qf[c][2] = (_Float16)q0.z;  qf[c][3] = (_Float16)q0.w;
    qf[c][4] = (_Float16)q1.x;  qf[c][5] = (_Float16)q1.y;
    qf[c][6] = (_Float16)q1.z;  qf[c][7] = (_Float16)q1.w;
    qf[c][8] = (_Float16)q2.x;  qf[c][9] = (_Float16)q2.y;
    qf[c][10] = (_Float16)q2.z; qf[c][11] = (_Float16)q2.w;
    qf[c][12] = (_Float16)q3.x; qf[c][13] = (_Float16)q3.y;
    qf[c][14] = (_Float16)q3.z; qf[c][15] = (_Float16)q3.w;
  }

  v8f acc[4];
#pragma unroll
  for (int t = 0; t < 4; ++t) acc[t] = vzero8();
  __syncthreads();

#pragma unroll 1
  for (int k0 = 0; k0 < 1024; k0 += 32) {
    // S = Q @ K^T for a 16x32 key block; K rows are contiguous in d
#pragma unroll
    for (int sub = 0; sub < 2; ++sub) {
      v8f sacc = vzero8();
      const int key = k0 + sub * 16 + r;
#pragma unroll
      for (int c = 0; c < 2; ++c) {
        const float* kp = Kb + base + (long)key * 64 + c * 32 + 8 * hi;
        const float4 k0v = *(const float4*)kp;
        const float4 k1v = *(const float4*)(kp + 4);
        const float4 k2v = *(const float4*)(kp + 16);
        const float4 k3v = *(const float4*)(kp + 20);
        v16h kf;
        kf[0] = (_Float16)k0v.x;  kf[1] = (_Float16)k0v.y;
        kf[2] = (_Float16)k0v.z;  kf[3] = (_Float16)k0v.w;
        kf[4] = (_Float16)k1v.x;  kf[5] = (_Float16)k1v.y;
        kf[6] = (_Float16)k1v.z;  kf[7] = (_Float16)k1v.w;
        kf[8] = (_Float16)k2v.x;  kf[9] = (_Float16)k2v.y;
        kf[10] = (_Float16)k2v.z; kf[11] = (_Float16)k2v.w;
        kf[12] = (_Float16)k3v.x; kf[13] = (_Float16)k3v.y;
        kf[14] = (_Float16)k3v.z; kf[15] = (_Float16)k3v.w;
        sacc = WMMA_F16(qf[c], kf, sacc);
      }
#pragma unroll
      for (int v = 0; v < 8; ++v)
        sS[v + 8 * hi][sub * 16 + r] = sacc[v] * 0.125f;  // DH^-0.5
    }
    __syncthreads();
    if (hi == 0) {
      float mx = sS[r][0];
#pragma unroll
      for (int k = 1; k < 32; ++k) mx = fmaxf(mx, sS[r][k]);
      const float mo = sM[r];
      const float mn = fmaxf(mo, mx);
      const float al = __expf(mo - mn);
      float ls = 0.f;
#pragma unroll
      for (int k = 0; k < 32; ++k) {
        const float p = __expf(sS[r][k] - mn);
        sS[r][k] = p;
        ls += p;
      }
      sM[r] = mn; sAl[r] = al; sL[r] = sL[r] * al + ls;
    }
    __syncthreads();
#pragma unroll
    for (int v = 0; v < 8; ++v) {
      const float al = sAl[v + 8 * hi];
#pragma unroll
      for (int t = 0; t < 4; ++t) acc[t][v] *= al;
    }
    v16h pf;
#pragma unroll
    for (int j = 0; j < 16; ++j) pf[j] = (_Float16)sS[r][a_k(hi, j)];
#pragma unroll
    for (int t = 0; t < 4; ++t) {
      const _Float16* vp = vT + ((long)bh * 64 + t * 16 + r) * 1024 + k0 + 8 * hi;
      const v8h b0 = *(const v8h*)vp;
      const v8h b1 = *(const v8h*)(vp + 16);
      v16h vf;
#pragma unroll
      for (int j = 0; j < 8; ++j) { vf[j] = b0[j]; vf[j + 8] = b1[j]; }
      acc[t] = WMMA_F16(pf, vf, acc[t]);
    }
    __syncthreads();
  }

#pragma unroll
  for (int v = 0; v < 8; ++v) {
    const float li = 1.f / sL[v + 8 * hi];
    const int so = st * 16 + v + 8 * hi;
#pragma unroll
    for (int t = 0; t < 4; ++t)
      ctx[base + (long)so * 64 + t * 16 + r] = acc[t][v] * li;
  }
}

// ---------------------------------------------------------------------------
// Small VALU kernels
// ---------------------------------------------------------------------------
// P_l_1[b,n,d] = sum_m s_adj[b,m,n] * P[b,m,d]  (contraction over M=8)
__global__ void pl1_kernel(const float* __restrict__ sadj,
                           const float* __restrict__ P,
                           float* __restrict__ out) {
  const int tid = blockIdx.x * blockDim.x + threadIdx.x;  // B*N*HID
  const int b = tid >> 18;
  const int n = (tid >> 8) & 1023;
  const int d = tid & 255;
  float s = 0.f;
#pragma unroll
  for (int m = 0; m < 8; ++m)
    s += sadj[((b * 8 + m) << 10) + n] * P[((b * 8 + m) << 8) + d];
  out[tid] = s;
}

// s1 = h@a1, s2 = h@a2 per (b,n); first two threads also produce rel_sc[2]
__global__ void s1s2_kernel(const float* __restrict__ hb,
                            const float* __restrict__ ga,   // gat_a[l] (768)
                            const float* __restrict__ rel,  // rel_emb[l] (2x256)
                            float* __restrict__ s1, float* __restrict__ s2,
                            float* __restrict__ relsc) {
  const int i = blockIdx.x * blockDim.x + threadIdx.x;  // B*N
  const float* row = hb + (long)i * 256;
  float d1 = 0.f, d2 = 0.f;
  for (int c = 0; c < 256; ++c) { d1 += row[c] * ga[c]; d2 += row[c] * ga[256 + c]; }
  s1[i] = d1; s2[i] = d2;
  if (i < 2) {
    float rs = 0.f;
    for (int c = 0; c < 256; ++c) rs += rel[i * 256 + c] * ga[512 + c];
    relsc[i] = rs;
  }
}

// P_agg[b,m,d] = sum_n s_adj[b,m,n] * H[b,n,d]
__global__ void pagg_kernel(const float* __restrict__ sadj,
                            const float* __restrict__ H,
                            float* __restrict__ out) {
  const int tid = blockIdx.x * blockDim.x + threadIdx.x;  // B*M*HID
  const int b = tid >> 11;
  const int m = (tid >> 8) & 7;
  const int d = tid & 255;
  const float* ar = sadj + ((b * 8 + m) << 10);
  float s = 0.f;
  for (int n = 0; n < 1024; ++n) s += ar[n] * H[(((long)b << 10) + n) * 256 + d];
  out[tid] = s;
}

__global__ void gru_kernel(const float* __restrict__ zp, const float* __restrict__ rp,
                           const float* __restrict__ t1, const float* __restrict__ t2,
                           const float* __restrict__ Pold, float* __restrict__ Pnew) {
  const int i = blockIdx.x * blockDim.x + threadIdx.x;
  const float z = 1.f / (1.f + __expf(-zp[i]));
  const float r = 1.f / (1.f + __expf(-rp[i]));
  const float hh = tanhf(t1[i] + r * t2[i]);
  Pnew[i] = (1.f - z) * Pold[i] + z * hh;
}

// In-place LayerNorm over last dim (256), one block per row
__global__ void ln_kernel(float* __restrict__ X, const float* __restrict__ g,
                          const float* __restrict__ b) {
  const int row = blockIdx.x;
  const int t = threadIdx.x;
  __shared__ float red[256];
  const float v = X[(long)row * 256 + t];
  red[t] = v;
  __syncthreads();
  for (int s = 128; s > 0; s >>= 1) { if (t < s) red[t] += red[t + s]; __syncthreads(); }
  const float mu = red[0] * (1.f / 256.f);
  __syncthreads();
  const float dl = v - mu;
  red[t] = dl * dl;
  __syncthreads();
  for (int s = 128; s > 0; s >>= 1) { if (t < s) red[t] += red[t + s]; __syncthreads(); }
  const float var = red[0] * (1.f / 256.f);
  X[(long)row * 256 + t] = dl * rsqrtf(var + 1e-5f) * g[t] + b[t];
}

// p_sim = mean over (b,m,k) of (P P^T with zeroed diagonal)
__global__ void psim_kernel(const float* __restrict__ P, float* __restrict__ out) {
  __shared__ float red[256];
  const int t = threadIdx.x;
  float accv = 0.f;
  for (int item = t; item < 512; item += 256) {
    const int b = item >> 6;
    const int m = (item >> 3) & 7;
    const int k = item & 7;
    if (m != k) {
      const float* pm = P + ((b * 8 + m) << 8);
      const float* pk = P + ((b * 8 + k) << 8);
      float d = 0.f;
      for (int c = 0; c < 256; ++c) d += pm[c] * pk[c];
      accv += d;
    }
  }
  red[t] = accv;
  __syncthreads();
  for (int s = 128; s > 0; s >>= 1) { if (t < s) red[t] += red[t + s]; __syncthreads(); }
  if (t == 0) out[0] = red[0] * (1.f / 512.f);  // /(B*M*M)
}

// ---------------------------------------------------------------------------
// Host-side pipeline
// ---------------------------------------------------------------------------
extern "C" void kernel_launch(void* const* d_in, const int* in_sizes, int n_in,
                              void* d_out, int out_size, void* d_ws, size_t ws_size,
                              hipStream_t stream) {
  const float* x     = (const float*)d_in[0];
  const int*   adjm  = (const int*)  d_in[1];
  const int*   smask = (const int*)  d_in[2];
  const float* sfeat = (const float*)d_in[3];
  const float* sadj  = (const float*)d_in[4];
  // d_in[5] speaker_id: unused by the reference (init_way == 'global')
  const float* fc1_w = (const float*)d_in[6];
  const float* fc1_b = (const float*)d_in[7];
  const float* fc2_w = (const float*)d_in[8];
  const float* fc2_b = (const float*)d_in[9];
  const float* gat_W = (const float*)d_in[10];
  const float* gat_a = (const float*)d_in[11];
  const float* rel_e = (const float*)d_in[12];
  const float* gwz = (const float*)d_in[13];
  const float* guz = (const float*)d_in[14];
  const float* gwr = (const float*)d_in[15];
  const float* gur = (const float*)d_in[16];
  const float* gw  = (const float*)d_in[17];
  const float* gu  = (const float*)d_in[18];
  const float* wv = (const float*)d_in[19]; const float* bv = (const float*)d_in[20];
  const float* wk = (const float*)d_in[21]; const float* bk = (const float*)d_in[22];
  const float* wq = (const float*)d_in[23]; const float* bq = (const float*)d_in[24];
  const float* wo = (const float*)d_in[25]; const float* bo = (const float*)d_in[26];
  const float* lng = (const float*)d_in[27]; const float* lnb = (const float*)d_in[28];
  const float* out_w = (const float*)d_in[29];
  const float* out_b = (const float*)d_in[30];

  float* ws = (float*)d_ws;
  const long BNH = (long)8 * 1024 * 256;
  float* H   = ws; ws += BNH;
  float* Hn  = ws; ws += BNH;
  float* Pl1 = ws; ws += BNH;
  float* hb  = ws; ws += BNH;
  float* Agg = ws; ws += BNH;
  float* Vb  = ws; ws += BNH;
  float* Kb  = ws; ws += BNH;
  float* Qb  = ws; ws += BNH;
  float* ctx = ws; ws += BNH;
  const int BMH = 8 * 8 * 256;
  float* P    = ws; ws += BMH;
  float* Pn   = ws; ws += BMH;
  float* zp   = ws; ws += BMH;
  float* rp   = ws; ws += BMH;
  float* t1   = ws; ws += BMH;
  float* t2   = ws; ws += BMH;
  float* Pagg = ws; ws += BMH;
  float* s1 = ws; ws += 8 * 1024;
  float* s2 = ws; ws += 8 * 1024;
  float* relsc = ws; ws += 2;
  _Float16* hT = (_Float16*)ws; ws += BNH / 2;  // f16 [B,HID,N]
  _Float16* vT = (_Float16*)ws; ws += BNH / 2;  // f16 [32,64,1024]

  auto gemm = [&](const float* Aq, const float* Wq, const float* bqv,
                  const float* addq, float* Cq, int Mr, int Kd, int Nc, int actq) {
    dim3 g(Mr / 64, (Nc + 63) / 64);
    gemm_wmma_kernel<<<g, 128, 0, stream>>>(Aq, Wq, bqv, addq, Cq, Mr, Kd, Nc, actq);
  };

  gemm(x, fc1_w, fc1_b, nullptr, H, 8192, 1024, 256, 1);    // H = relu(x@fc1)
  gemm(sfeat, fc2_w, fc2_b, nullptr, P, 64, 1024, 256, 1);  // P = relu(sf@fc2)

  float* Hc = H; float* Hx = Hn; float* Pc = P; float* Px = Pn;
  for (int l = 0; l < 2; ++l) {
    const float* gatWl = gat_W + (long)l * 65536;
    const float* gatal = gat_a + (long)l * 768;
    const float* rell  = rel_e + (long)l * 512;

    pl1_kernel<<<(8 * 1024 * 256) / 256, 256, 0, stream>>>(sadj, Pc, Pl1);
    gemm(Hc, gatWl, nullptr, nullptr, hb, 8192, 256, 256, 0);  // h = H@W
    h_transpose_kernel<<<(8 * 1024 * 256) / 256, 256, 0, stream>>>(hb, hT);
    s1s2_kernel<<<(8 * 1024) / 256, 256, 0, stream>>>(hb, gatal, rell, s1, s2, relsc);
    gat_agg_kernel<<<8 * 64, 32, 0, stream>>>(hT, s1, s2, smask, adjm, relsc, Agg);

    gemm(Hc,  wv + (long)l * 65536, bv + l * 256, nullptr, Vb, 8192, 256, 256, 0);
    gemm(Pl1, wk + (long)l * 65536, bk + l * 256, nullptr, Kb, 8192, 256, 256, 0);
    gemm(Agg, wq + (long)l * 65536, bq + l * 256, nullptr, Qb, 8192, 256, 256, 0);
    v_transpose_kernel<<<(8 * 1024 * 256) / 256, 256, 0, stream>>>(Vb, vT);
    mha_kernel<<<32 * 64, 32, 0, stream>>>(Qb, Kb, vT, ctx);
    gemm(ctx, wo + (long)l * 65536, bo + l * 256, Hc, Hx, 8192, 256, 256, 0);
    ln_kernel<<<8192, 256, 0, stream>>>(Hx, lng + l * 256, lnb + l * 256);

    pagg_kernel<<<(8 * 8 * 256) / 256, 256, 0, stream>>>(sadj, Hc, Pagg);
    gemm(Pc,   gwz + (long)l * 65536, nullptr, nullptr, zp, 64, 256, 256, 0);
    gemm(Pagg, guz + (long)l * 65536, nullptr, zp,      zp, 64, 256, 256, 0);
    gemm(Pc,   gwr + (long)l * 65536, nullptr, nullptr, rp, 64, 256, 256, 0);
    gemm(Pagg, gur + (long)l * 65536, nullptr, rp,      rp, 64, 256, 256, 0);
    gemm(Pc,   gw  + (long)l * 65536, nullptr, nullptr, t1, 64, 256, 256, 0);
    gemm(Pagg, gu  + (long)l * 65536, nullptr, nullptr, t2, 64, 256, 256, 0);
    gru_kernel<<<(8 * 8 * 256) / 256, 256, 0, stream>>>(zp, rp, t1, t2, Pc, Px);

    float* tf = Hc; Hc = Hx; Hx = tf;
    tf = Pc; Pc = Px; Px = tf;
  }

  gemm(Hc, out_w, out_b, nullptr, (float*)d_out, 8192, 256, 7, 0);   // logits
  psim_kernel<<<1, 256, 0, stream>>>(Pc, (float*)d_out + 8192 * 7);  // p_sim

  (void)in_sizes; (void)n_in; (void)out_size; (void)ws_size;
}
